// Attention_24206435680829
// MI455X (gfx1250) — matrix operand order repaired and checked by reading, not compiled, not run on hardware
//
#include <hip/hip_runtime.h>

// Problem constants from the reference
constexpr int T = 4096;
constexpr int B = 32;
constexpr int H = 512;
constexpr int R = 512;

typedef __attribute__((ext_vector_type(16))) _Float16 v16h;
typedef __attribute__((ext_vector_type(8)))  float    v8f;
typedef __attribute__((ext_vector_type(4)))  float    v4f;   // native vector: OK for nontemporal builtins

// ---------------------------------------------------------------------------
// tanh: prefer native v_tanh_f32 (1 TRANS op) if the toolchain exposes it,
// else exp2-based (2 TRANS + few VALU). Memory floor is ~11us; native tanh
// keeps VALU time at/below that.
// ---------------------------------------------------------------------------
__device__ __forceinline__ float fast_tanh(float x) {
#if defined(__has_builtin) && __has_builtin(__builtin_amdgcn_tanhf)
    return __builtin_amdgcn_tanhf(x);
#else
    // tanh(x) = 1 - 2/(exp(2x)+1)
    float e = __expf(2.0f * x);
    return 1.0f - 2.0f / (e + 1.0f);
#endif
}

// ---------------------------------------------------------------------------
// K1: rec[b,h] = sum_r rnn_state[b,r] * W_rec[h,r]
// One wave per 16x16 output tile; K-loop of 32 using v_wmma_f32_16x16x32_f16.
// A tile = rnn_state rows (M=b, K=r), B tile[k,n] = W_rec[h0+n, k0+k]
// (each lane reads 16 contiguous floats of one W_rec row -> its B column).
// A/B use contiguous K order (lane half h holds K=16h..16h+15) in both; D per ISA doc (wave32).
// ---------------------------------------------------------------------------
__global__ __launch_bounds__(32)
void rec_gemm_wmma(const float* __restrict__ S,   // (B,R) rnn_state
                   const float* __restrict__ W,   // (H,R) W_rec
                   float* __restrict__ rec)       // (B,H)
{
    const int lane = threadIdx.x;        // 0..31
    const int tile = blockIdx.x;         // 64 tiles = 2 (M) x 32 (N)
    const int b0 = (tile & 1) * 16;      // M tile base (B dim)
    const int h0 = (tile >> 1) * 16;     // N tile base (H dim)

    const int lo    = (lane < 16);
    const int aRow  = b0 + (lane & 15);
    const int aKofs = lo ? 0 : 16;       // A: lane holds K=aKofs..aKofs+15 of its row
    const int bCol  = h0 + (lane & 15);
    const int bKofs = lo ? 0 : 16;       // B: lane holds K=bKofs..bKofs+15 of its column

    v8f acc = {};
    for (int k0 = 0; k0 < R; k0 += 32) {
        v16h a, bm;
        const float* ap = S + (size_t)aRow * R + k0 + aKofs;
        #pragma unroll
        for (int j = 0; j < 16; ++j) {
            a[j]     = (_Float16)ap[j];
        }
        const float* bp = W + (size_t)bCol * R + k0 + bKofs;
        #pragma unroll
        for (int j = 0; j < 16; ++j) bm[j] = (_Float16)bp[j];

        acc = __builtin_amdgcn_wmma_f32_16x16x32_f16(
            /*neg_a=*/false, a, /*neg_b=*/false, bm,
            /*c_mod=*/(short)0, acc, /*reuse_a=*/false, /*reuse_b=*/false);
    }

    // D layout: VGPR i -> (M = i or 8+i, N = lane&15)
    const int n     = h0 + (lane & 15);
    const int mbase = b0 + (lo ? 0 : 8);
    #pragma unroll
    for (int i = 0; i < 8; ++i)
        rec[(size_t)(mbase + i) * H + n] = acc[i];
}

// ---------------------------------------------------------------------------
// K2: scores[t,b] = sum_h tanh(enc[t,b,h] + rec[b,h]) * w_score[h] + b0 + mask
// Block = 256 threads (8 waves), blockIdx.y = b, each wave handles one t.
// enc row (512 floats, contiguous) read as 128-bit vectors with NON-TEMPORAL
// hint: enc is a 256 MB single-use stream and must not displace the reused
// data (rec, w_score, scores) from WGP$/L2. rec[b,:] and w_score in LDS.
// ---------------------------------------------------------------------------
__global__ __launch_bounds__(256)
void score_kernel(const float* __restrict__ enc,     // (T,B,H)
                  const float* __restrict__ rec,     // (B,H)
                  const float* __restrict__ wsc,     // (H)
                  const float* __restrict__ mask,    // (T,B)
                  const float* __restrict__ bsc,     // (1)
                  float* __restrict__ scores)        // (T,B)
{
    __shared__ float s_rec[H];
    __shared__ float s_w[H];

    const int b   = blockIdx.y;
    const int tid = threadIdx.x;
    for (int i = tid; i < H; i += 256) {
        s_rec[i] = rec[(size_t)b * H + i];
        s_w[i]   = wsc[i];
    }
    __syncthreads();

    const int wave = tid >> 5;
    const int lane = tid & 31;
    const int t    = blockIdx.x * 8 + wave;

    const v4f* row = (const v4f*)(enc + ((size_t)t * B + b) * H);

    float acc = 0.0f;
    #pragma unroll
    for (int i = 0; i < 4; ++i) {
        const int i4 = i * 32 + lane;        // 128 x v4f = 512 floats
        const v4f e = __builtin_nontemporal_load(row + i4);  // global_load_b128 TH=NT
        const int h = i4 * 4;
        acc += fast_tanh(e.x + s_rec[h    ]) * s_w[h    ];
        acc += fast_tanh(e.y + s_rec[h + 1]) * s_w[h + 1];
        acc += fast_tanh(e.z + s_rec[h + 2]) * s_w[h + 2];
        acc += fast_tanh(e.w + s_rec[h + 3]) * s_w[h + 3];
    }

    #pragma unroll
    for (int off = 16; off > 0; off >>= 1)
        acc += __shfl_xor(acc, off, 32);

    if (lane == 0) {
        const float mk = __builtin_nontemporal_load(mask + (size_t)t * B + b);
        scores[(size_t)t * B + b] = acc + bsc[0] + mk;   // RT store: K3 re-reads from L2
    }
}

// ---------------------------------------------------------------------------
// K3: softmax over T (axis 0) per column b. One block (256 thr) per b,
// 16 t-values per thread held in registers.
// ---------------------------------------------------------------------------
__global__ __launch_bounds__(256)
void softmax_T(const float* __restrict__ scores,  // (T,B)
               float* __restrict__ out)           // (T,B)
{
    constexpr int PER = T / 256;   // 16
    const int b   = blockIdx.x;
    const int tid = threadIdx.x;
    const int wave = tid >> 5, lane = tid & 31;

    __shared__ float red[8];
    float vals[PER];

    float m = -3.0e38f;
    #pragma unroll
    for (int i = 0; i < PER; ++i) {
        const int t = tid + i * 256;
        const float s = scores[(size_t)t * B + b];
        vals[i] = s;
        m = fmaxf(m, s);
    }
    #pragma unroll
    for (int off = 16; off > 0; off >>= 1)
        m = fmaxf(m, __shfl_xor(m, off, 32));
    if (lane == 0) red[wave] = m;
    __syncthreads();
    if (tid == 0) {
        float mm = red[0];
        for (int w = 1; w < 8; ++w) mm = fmaxf(mm, red[w]);
        red[0] = mm;
    }
    __syncthreads();
    m = red[0];
    __syncthreads();

    float sum = 0.0f;
    #pragma unroll
    for (int i = 0; i < PER; ++i) {
        vals[i] = __expf(vals[i] - m);
        sum += vals[i];
    }
    #pragma unroll
    for (int off = 16; off > 0; off >>= 1)
        sum += __shfl_xor(sum, off, 32);
    if (lane == 0) red[wave] = sum;
    __syncthreads();
    if (tid == 0) {
        float ss = 0.0f;
        for (int w = 0; w < 8; ++w) ss += red[w];
        red[0] = ss;
    }
    __syncthreads();
    const float inv = 1.0f / red[0];

    #pragma unroll
    for (int i = 0; i < PER; ++i) {
        const int t = tid + i * 256;
        out[(size_t)t * B + b] = vals[i] * inv;
    }
}

// ---------------------------------------------------------------------------
extern "C" void kernel_launch(void* const* d_in, const int* in_sizes, int n_in,
                              void* d_out, int out_size, void* d_ws, size_t ws_size,
                              hipStream_t stream) {
    const float* enc   = (const float*)d_in[0];   // (T,B,H)
    const float* mask  = (const float*)d_in[1];   // (T,B)
    const float* rnn   = (const float*)d_in[2];   // (B,R)
    // d_in[3] = prev_att_weights, unused by the reference
    const float* W_rec = (const float*)d_in[4];   // (H,R)
    const float* w_sc  = (const float*)d_in[5];   // (H)
    const float* b_sc  = (const float*)d_in[6];   // (1)
    float* out = (float*)d_out;                   // (T,B)

    float* ws_rec    = (float*)d_ws;              // B*H floats = 64 KB
    float* ws_scores = ws_rec + (size_t)B * H;    // T*B floats = 512 KB

    // K1: rec = rnn_state @ W_rec^T via WMMA (2x32 tiles, 1 wave each)
    rec_gemm_wmma<<<dim3(64), dim3(32), 0, stream>>>(rnn, W_rec, ws_rec);

    // K2: streaming tanh-dot (memory-bound: 256 MB @ 23.3 TB/s ~ 11 us)
    score_kernel<<<dim3(T / 8, B), dim3(256), 0, stream>>>(
        enc, ws_rec, w_sc, mask, b_sc, ws_scores);

    // K3: softmax over T
    softmax_T<<<dim3(B), dim3(256), 0, stream>>>(ws_scores, out);
}
